// PointNetV11Encoder_74028056314050
// MI455X (gfx1250) — matrix operand order repaired and checked by reading, not compiled, not run on hardware
//
#include <hip/hip_runtime.h>

// ---------------------------------------------------------------------------
// PointNet++ SA encoder for MI455X (gfx1250, wave32, WMMA bf16)
// ---------------------------------------------------------------------------

typedef __bf16 bf16_t;
typedef __attribute__((ext_vector_type(16))) __bf16 v16bf;
typedef __attribute__((ext_vector_type(8)))  __bf16 v8bf;
typedef __attribute__((ext_vector_type(8)))  float  v8f;

static constexpr int NB   = 8;      // batches
static constexpr int NP   = 8192;   // points per batch
static constexpr int MC   = 1024;   // centers (G_N)
static constexpr int KG   = 32;     // group size (G_K)
static constexpr int RTOT = NB * MC * KG;   // 262144 rows
static constexpr float BN_EPS = 1e-5f;
static constexpr float RINV   = 1.0f / (float)RTOT;

// ---------------------------------------------------------------------------
// 1) Farthest point sampling: one workgroup per batch, points + dists in LDS
// ---------------------------------------------------------------------------
__global__ __launch_bounds__(1024)
void fps_kernel(const float* __restrict__ p, int* __restrict__ u_ce,
                float* __restrict__ p_ce) {
  extern __shared__ float smem[];
  float* px   = smem;
  float* py   = px + NP;
  float* pz   = py + NP;
  float* dist = pz + NP;
  float* rv   = dist + NP;          // 1024 reduction values
  int*   ri   = (int*)(rv + 1024);  // 1024 reduction indices
  int*   sel  = ri + 1024;          // 1024 selected indices
  __shared__ int s_far;

  const int b = blockIdx.x;
  const int t = threadIdx.x;
  const float* pb = p + (size_t)b * NP * 3;

  for (int i = t; i < NP; i += 1024) {
    px[i] = pb[i * 3 + 0];
    py[i] = pb[i * 3 + 1];
    pz[i] = pb[i * 3 + 2];
    dist[i] = 1e10f;
  }
  if (t == 0) s_far = 0;
  __syncthreads();

  for (int it = 0; it < MC; ++it) {
    const int far = s_far;
    if (t == 0) sel[it] = far;
    const float cx = px[far], cy = py[far], cz = pz[far];
    float bv = -1.0f; int bi = 0;
    for (int i = t; i < NP; i += 1024) {
      float dx = px[i] - cx, dy = py[i] - cy, dz = pz[i] - cz;
      float d  = dx * dx + dy * dy + dz * dz;
      float dd = fminf(dist[i], d);
      dist[i] = dd;
      if (dd > bv) { bv = dd; bi = i; }   // strict > keeps smallest index
    }
    rv[t] = bv; ri[t] = bi;
    __syncthreads();
    for (int s = 512; s > 0; s >>= 1) {
      if (t < s) {
        float v2 = rv[t + s]; int i2 = ri[t + s];
        if (v2 > rv[t] || (v2 == rv[t] && i2 < ri[t])) { rv[t] = v2; ri[t] = i2; }
      }
      __syncthreads();
    }
    if (t == 0) s_far = ri[0];
    __syncthreads();
  }

  for (int m = t; m < MC; m += 1024) {
    int idx = sel[m];
    u_ce[b * MC + m] = idx;
    float* o = p_ce + ((size_t)b * MC + m) * 3;
    o[0] = px[idx]; o[1] = py[idx]; o[2] = pz[idx];
  }
}

// ---------------------------------------------------------------------------
// 2) Ball query: one wave32 per center, ballot + prefix-popcount compaction
// ---------------------------------------------------------------------------
__global__ __launch_bounds__(256)
void ballq_kernel(const float* __restrict__ p, const float* __restrict__ p_ce,
                  int* __restrict__ u_ne) {
  const float r2 = 0.1f * 0.1f;
  const int lane = threadIdx.x & 31;
  const int wid  = blockIdx.x * (blockDim.x >> 5) + (threadIdx.x >> 5);
  const int b = wid >> 10;                 // wid in [0, 8192)
  const float* pb = p + (size_t)b * NP * 3;
  const float* c  = p_ce + (size_t)wid * 3;
  const float cx = c[0], cy = c[1], cz = c[2];
  int* out = u_ne + (size_t)wid * KG;

  int cnt = 0;
  for (int base = 0; base < NP && cnt < KG; base += 32) {
    const int n = base + lane;
    float dx = pb[n * 3 + 0] - cx;
    float dy = pb[n * 3 + 1] - cy;
    float dz = pb[n * 3 + 2] - cz;
    float d  = dx * dx + dy * dy + dz * dz;
    unsigned mask = __builtin_amdgcn_ballot_w32(d < r2);
    if (d < r2) {
      int slot = cnt + __popc(mask & ((1u << lane) - 1u));
      if (slot < KG) out[slot] = n;
    }
    cnt += __popc(mask);
  }
  if (lane == 0) {
    if (cnt == 0) {
      for (int k2 = 0; k2 < KG; ++k2) out[k2] = NP - 1;
    } else if (cnt < KG) {
      int f = out[0];
      for (int k2 = cnt; k2 < KG; ++k2) out[k2] = f;
    }
  }
}

// ---------------------------------------------------------------------------
// 3) Gather + concat into bf16 activation matrix X [RTOT x 32] (pad 16..31=0)
// ---------------------------------------------------------------------------
__global__ __launch_bounds__(256)
void build_x_kernel(const float* __restrict__ f, const float* __restrict__ p,
                    const float* __restrict__ p_ce, const int* __restrict__ u_ne,
                    bf16_t* __restrict__ X) {
  const int row = blockIdx.x * 256 + threadIdx.x;   // RTOT rows
  const int bm  = row >> 5;
  const int b   = bm >> 10;
  const int idx = u_ne[row];
  const float* fr = f + ((size_t)b * NP + idx) * 13;
  const float* pr = p + ((size_t)b * NP + idx) * 3;
  const float* c  = p_ce + (size_t)bm * 3;
  bf16_t* xr = X + (size_t)row * 32;
#pragma unroll
  for (int i = 0; i < 13; ++i) xr[i] = (bf16_t)fr[i];
  xr[13] = (bf16_t)(pr[0] - c[0]);
  xr[14] = (bf16_t)(pr[1] - c[1]);
  xr[15] = (bf16_t)(pr[2] - c[2]);
#pragma unroll
  for (int i = 16; i < 32; ++i) xr[i] = (bf16_t)0.0f;
}

// ---------------------------------------------------------------------------
// 4) Weight conversion (f32 -> bf16, W0 zero-padded 16->32) + zero BN stats
// ---------------------------------------------------------------------------
__global__ __launch_bounds__(256)
void prep_kernel(const float* __restrict__ W0, const float* __restrict__ W1a,
                 const float* __restrict__ W1b, const float* __restrict__ W2,
                 bf16_t* __restrict__ w0p, bf16_t* __restrict__ w1a,
                 bf16_t* __restrict__ w1b, bf16_t* __restrict__ w2,
                 float* __restrict__ stats) {
  const int i = blockIdx.x * 256 + threadIdx.x;
  if (i < 896) stats[i] = 0.0f;
  if (i < 64 * 32) {
    int o = i >> 5, c = i & 31;
    w0p[i] = (bf16_t)((c < 16) ? W0[o * 16 + c] : 0.0f);
  }
  if (i < 128 * 64)  w1a[i] = (bf16_t)W1a[i];
  if (i < 128 * 128) w1b[i] = (bf16_t)W1b[i];
  if (i < 128 * 64)  w2[i]  = (bf16_t)W2[i];
}

// ---------------------------------------------------------------------------
// 5) WMMA GEMM: Y[R x COUT] = X[R x CIN] * W[COUT x CIN]^T  (bf16 in, f32 acc)
//    8 waves/block, 16 rows/wave -> 128 rows per workgroup.
// ---------------------------------------------------------------------------
template <int CIN, int COUT>
__global__ __launch_bounds__(256)
void gemm_bf16_kernel(const bf16_t* __restrict__ X, const bf16_t* __restrict__ W,
                      bf16_t* __restrict__ Y) {
  constexpr int KF = CIN / 32;    // K fragments
  constexpr int NT = COUT / 16;   // N tiles
  const int wave = threadIdx.x >> 5;
  const int lane = threadIdx.x & 31;
  const int half = lane >> 4;
  const int r    = lane & 15;
  const int row0 = blockIdx.x * 128 + wave * 16;

  // A fragments: lanes 0-15 -> K {0..7,16..23}; lanes 16-31 -> K {8..15,24..31}
  v16bf afr[KF];
  const bf16_t* xrow = X + (size_t)(row0 + r) * CIN;
#pragma unroll
  for (int kk = 0; kk < KF; ++kk) {
    v8bf c0 = *(const v8bf*)(xrow + kk * 32 + (half ? 8 : 0));
    v8bf c1 = *(const v8bf*)(xrow + kk * 32 + (half ? 24 : 16));
    v16bf a{};
#pragma unroll
    for (int i = 0; i < 8; ++i) { a[i] = c0[i]; a[i + 8] = c1[i]; }
    afr[kk] = a;
  }

#pragma unroll
  for (int t = 0; t < NT; ++t) {
    v8f acc = {};
    const bf16_t* wrow = W + (size_t)(t * 16 + r) * CIN;  // column n of B
#pragma unroll
    for (int kk = 0; kk < KF; ++kk) {
      // B: column-per-lane, lanes 0-15 -> K {0..7,16..23}; lanes 16-31 -> K {8..15,24..31}
      v8bf b0 = *(const v8bf*)(wrow + kk * 32 + half * 8);
      v8bf b1 = *(const v8bf*)(wrow + kk * 32 + 16 + half * 8);
      v16bf bf{};
#pragma unroll
      for (int i = 0; i < 8; ++i) { bf[i] = b0[i]; bf[i + 8] = b1[i]; }
      acc = __builtin_amdgcn_wmma_f32_16x16x32_bf16(
          false, afr[kk], false, bf, (short)0, acc, false, false);
    }
    // D layout: lane(0-15)=N, vgpr j = M j; lane(16-31): M = j+8
    const int n     = t * 16 + r;
    const int mrow0 = row0 + (half ? 8 : 0);
#pragma unroll
    for (int j = 0; j < 8; ++j)
      Y[(size_t)(mrow0 + j) * COUT + n] = (bf16_t)acc[j];
  }
}

// ---------------------------------------------------------------------------
// 6) BN stats: per-channel sum / sumsq over all RTOT rows
// ---------------------------------------------------------------------------
template <int C>
__global__ void bnstats_kernel(const bf16_t* __restrict__ Y,
                               float* __restrict__ sum, float* __restrict__ sq) {
  const int c = threadIdx.x;             // blockDim == C
  const size_t row0 = (size_t)blockIdx.x * 256;
  float s = 0.0f, q = 0.0f;
  for (int r = 0; r < 256; ++r) {
    float v = (float)Y[(row0 + r) * C + c];
    s += v; q += v * v;
  }
  atomicAdd(&sum[c], s);
  atomicAdd(&sq[c], q);
}

// ---------------------------------------------------------------------------
// 7) BN apply (+ optional ReLU), in place
// ---------------------------------------------------------------------------
template <int C, bool RELU>
__global__ __launch_bounds__(256)
void bnapply_kernel(bf16_t* __restrict__ Y, const float* __restrict__ sum,
                    const float* __restrict__ sq, const float* __restrict__ g,
                    const float* __restrict__ bb) {
  const size_t i = (size_t)blockIdx.x * 256 + threadIdx.x;
  const int c = (int)(i % C);
  float m = sum[c] * RINV;
  float v = sq[c] * RINV - m * m;
  float sc = g[c] * rsqrtf(v + BN_EPS);
  float y = ((float)Y[i] - m) * sc + bb[c];
  if (RELU) y = fmaxf(y, 0.0f);
  Y[i] = (bf16_t)y;
}

// ---------------------------------------------------------------------------
// 8) Final: y = relu(bn1b(Y1b) + bn2(Y2)); f_ce = max over k (32 rows)
// ---------------------------------------------------------------------------
__global__ __launch_bounds__(128)
void final_kernel(const bf16_t* __restrict__ Y1b, const bf16_t* __restrict__ Y2,
                  const float* __restrict__ s1s, const float* __restrict__ s1q,
                  const float* __restrict__ g1, const float* __restrict__ b1,
                  const float* __restrict__ s2s, const float* __restrict__ s2q,
                  const float* __restrict__ g2, const float* __restrict__ b2,
                  float* __restrict__ f_ce) {
  const int c = threadIdx.x;             // 128 channels
  const int grp = blockIdx.x;            // 8192 groups (b,m)
  float m1 = s1s[c] * RINV, v1 = s1q[c] * RINV - m1 * m1;
  float sc1 = g1[c] * rsqrtf(v1 + BN_EPS), o1 = b1[c] - m1 * sc1;
  float m2 = s2s[c] * RINV, v2 = s2q[c] * RINV - m2 * m2;
  float sc2 = g2[c] * rsqrtf(v2 + BN_EPS), o2 = b2[c] - m2 * sc2;
  const size_t base = (size_t)grp * KG;
  float mx = 0.0f;   // relu output >= 0, so 0 is a safe identity
  for (int k = 0; k < KG; ++k) {
    float y = ((float)Y1b[(base + k) * 128 + c] * sc1 + o1) +
              ((float)Y2[(base + k) * 128 + c] * sc2 + o2);
    mx = fmaxf(mx, fmaxf(y, 0.0f));
  }
  f_ce[(size_t)grp * 128 + c] = mx;
}

// ---------------------------------------------------------------------------
// Host-side orchestration
// ---------------------------------------------------------------------------
static inline size_t align256(size_t x) { return (x + 255) & ~(size_t)255; }

extern "C" void kernel_launch(void* const* d_in, const int* in_sizes, int n_in,
                              void* d_out, int out_size, void* d_ws, size_t ws_size,
                              hipStream_t stream) {
  const float* f   = (const float*)d_in[0];
  const float* p   = (const float*)d_in[1];
  const float* W0  = (const float*)d_in[2];
  const float* g0  = (const float*)d_in[3];
  const float* b0  = (const float*)d_in[4];
  const float* W1a = (const float*)d_in[5];
  const float* g1a = (const float*)d_in[6];
  const float* b1a = (const float*)d_in[7];
  const float* W1b = (const float*)d_in[8];
  const float* g1b = (const float*)d_in[9];
  const float* b1b = (const float*)d_in[10];
  const float* W2  = (const float*)d_in[11];
  const float* g2  = (const float*)d_in[12];
  const float* b2  = (const float*)d_in[13];

  float* out_fce = (float*)d_out;                        // [8,1024,128]
  float* out_pce = out_fce + (size_t)NB * MC * 128;      // [8,1024,3]

  // Workspace layout
  char* ws = (char*)d_ws;
  size_t cur = 0;
  int*    u_ce  = (int*)(ws + cur);  cur = align256(cur + (size_t)NB * MC * 4);
  int*    u_ne  = (int*)(ws + cur);  cur = align256(cur + (size_t)NB * MC * KG * 4);
  float*  stats = (float*)(ws + cur); cur = align256(cur + 896 * 4);
  bf16_t* w0p   = (bf16_t*)(ws + cur); cur = align256(cur + (size_t)64 * 32 * 2);
  bf16_t* w1a   = (bf16_t*)(ws + cur); cur = align256(cur + (size_t)128 * 64 * 2);
  bf16_t* w1b   = (bf16_t*)(ws + cur); cur = align256(cur + (size_t)128 * 128 * 2);
  bf16_t* w2    = (bf16_t*)(ws + cur); cur = align256(cur + (size_t)128 * 64 * 2);
  bf16_t* X     = (bf16_t*)(ws + cur); cur = align256(cur + (size_t)RTOT * 32 * 2);
  bf16_t* Y0    = (bf16_t*)(ws + cur); cur = align256(cur + (size_t)RTOT * 64 * 2);
  bf16_t* Y1a   = (bf16_t*)(ws + cur); cur = align256(cur + (size_t)RTOT * 128 * 2);
  bf16_t* Y1b   = (bf16_t*)(ws + cur); cur = align256(cur + (size_t)RTOT * 128 * 2);
  bf16_t* Y2    = (bf16_t*)(ws + cur); cur = align256(cur + (size_t)RTOT * 128 * 2);

  float* st0s  = stats + 0,   *st0q  = stats + 64;
  float* st1as = stats + 128, *st1aq = stats + 256;
  float* st1bs = stats + 384, *st1bq = stats + 512;
  float* st2s  = stats + 640, *st2q  = stats + 768;

  // Weight convert + zero stats
  prep_kernel<<<64, 256, 0, stream>>>(W0, W1a, W1b, W2, w0p, w1a, w1b, w2, stats);

  // FPS (dynamic LDS: 4*8192 + 1024 floats + 2048 ints = 140 KB, fine in 320 KB WGP LDS)
  size_t fps_smem = ((size_t)4 * NP + 1024) * 4 + (size_t)2 * 1024 * 4;
  fps_kernel<<<NB, 1024, fps_smem, stream>>>(p, u_ce, out_pce);

  // Ball query: 8192 centers, 8 waves/block
  ballq_kernel<<<NB * MC / 8, 256, 0, stream>>>(p, out_pce, u_ne);

  // Build X (bf16, 32 padded channels)
  build_x_kernel<<<RTOT / 256, 256, 0, stream>>>(f, p, out_pce, u_ne, X);

  const int GG = RTOT / 128;  // gemm grid (2048)

  // Layer 0: X(32) -> Y0(64); bn0 + relu in place -> H
  gemm_bf16_kernel<32, 64><<<GG, 256, 0, stream>>>(X, w0p, Y0);
  bnstats_kernel<64><<<RTOT / 256, 64, 0, stream>>>(Y0, st0s, st0q);
  bnapply_kernel<64, true><<<(RTOT * 64) / 256, 256, 0, stream>>>(Y0, st0s, st0q, g0, b0);

  // Branch 1: H(64) -> Y1a(128); bn1a + relu in place -> A1; A1(128) -> Y1b(128)
  gemm_bf16_kernel<64, 128><<<GG, 256, 0, stream>>>(Y0, w1a, Y1a);
  bnstats_kernel<128><<<RTOT / 256, 128, 0, stream>>>(Y1a, st1as, st1aq);
  bnapply_kernel<128, true><<<(RTOT * 128) / 256, 256, 0, stream>>>(Y1a, st1as, st1aq, g1a, b1a);
  gemm_bf16_kernel<128, 128><<<GG, 256, 0, stream>>>(Y1a, w1b, Y1b);

  // Branch 2: H(64) -> Y2(128)
  gemm_bf16_kernel<64, 128><<<GG, 256, 0, stream>>>(Y0, w2, Y2);

  // Stats for the two pre-BN branches
  bnstats_kernel<128><<<RTOT / 256, 128, 0, stream>>>(Y1b, st1bs, st1bq);
  bnstats_kernel<128><<<RTOT / 256, 128, 0, stream>>>(Y2, st2s, st2q);

  // Fused bn1b + bn2 + add + relu + max-over-k
  final_kernel<<<NB * MC, 128, 0, stream>>>(Y1b, Y2, st1bs, st1bq, g1b, b1b,
                                            st2s, st2q, g2, b2, out_fce);
}
